// node_interaction_90529320665784
// MI455X (gfx1250) — compile-verified, matrix operand order checked by reading, not run on hardware
//
#include <hip/hip_runtime.h>

// Problem constants (from reference; edge_num input ignored — it's fixed at 12000)
#define EN 12000            // edge_num
#define DD (49 * 128)       // 6272
#define HH 256              // hidden
#define TWO_E (2 * EN)

typedef __attribute__((ext_vector_type(16))) __bf16 v16bf;
typedef __attribute__((ext_vector_type(8)))  __bf16 v8bf;
typedef __attribute__((ext_vector_type(8)))  float  v8f;
typedef __attribute__((ext_vector_type(4)))  unsigned int v4u;
typedef __attribute__((ext_vector_type(8)))  int v8i;
typedef __attribute__((ext_vector_type(4)))  int v4i;

__device__ __forceinline__ unsigned int pack2bf(float a, float b) {
    union { __bf16 h[2]; unsigned int u; } p;
    p.h[0] = (__bf16)a;
    p.h[1] = (__bf16)b;
    return p.u;
}

// LDS tile layout: [row][k] with k contiguous, 32 bf16 (64B) per row.
// CDNA5 16-bit A/B fragment (16x32 / 32x16): lane = row%16 + 16*kgroup.
//   VGPR0..3 : K = kgroup*8 + 0..7      (contiguous 16 bytes)
//   VGPR4..7 : K = 16 + kgroup*8 + 0..7 (contiguous 16 bytes)
// -> two ds_load_b128 per fragment.
__device__ __forceinline__ v16bf frag_ld(const __bf16* p, int row, int lane) {
    union { v16bf v; v8bf h[2]; } f;
    const int kb = ((lane >> 4) & 1) * 8;
    const __bf16* q = p + row * 32;
    f.h[0] = *(const v8bf*)(q + kb);
    f.h[1] = *(const v8bf*)(q + 16 + kb);
    return f.v;
}

// Fast SiLU: x * rcp(1 + e^-x) -> v_exp_f32 + v_rcp_f32, no div chain.
__device__ __forceinline__ float silu_f(float x) {
    float e = __expf(-x);
    return x * __builtin_amdgcn_rcpf(1.0f + e);
}

// ---------------------------------------------------------------------------
// Tensor Data Mover: DMA a 2D bf16 tile (tile_dim0=32 elems x tile_dim1=64
// rows, row pitch 256 elems) from global to LDS. Rows beyond rows_valid read
// as zero (tensor-dim OOB), which is exactly our ragged-edge padding.
// D# layout per CDNA5 ISA ch.8 (data_size=1 -> 2 bytes/elem, type=2).
// ---------------------------------------------------------------------------
__device__ __forceinline__ void tdm_load_tile_64x32(unsigned lds_byte_addr,
                                                    const void* gaddr,
                                                    unsigned rows_valid) {
    unsigned long long ga = (unsigned long long)(uintptr_t)gaddr;
    v4u g0;
    g0[0] = 1u;                                   // count=1 (valid user D#)
    g0[1] = lds_byte_addr;                        // lds_addr  [63:32]
    g0[2] = (unsigned)(ga & 0xFFFFFFFFu);         // global_addr[31:0]
    g0[3] = (unsigned)((ga >> 32) & 0x01FFFFFFu)  // global_addr[56:32]
          | (2u << 30);                           // type=2 ("image")
    v8i g1;
    g1[0] = (int)(1u << 16);                      // data_size=1 (2B); mask=0
    g1[1] = (int)(256u << 16);                    // tensor_dim0 = 256
    g1[2] = (int)((rows_valid & 0xFFFFu) << 16);  // tensor_dim1 = valid rows
    g1[3] = (int)(32u << 16);                     // tile_dim0 = 32 elems (64B)
    g1[4] = (int)64;                              // tile_dim1 = 64 rows
    g1[5] = (int)256;                             // tensor_dim0_stride = 256
    g1[6] = 0;
    g1[7] = 0;
    v4i z4 = (v4i){0, 0, 0, 0};                   // 2D tile: groups 2/3 unused
#if defined(__clang_major__) && (__clang_major__ >= 23)
    v8i z8 = (v8i){};
    __builtin_amdgcn_tensor_load_to_lds(g0, g1, z4, z4, z8, 0);
#else
    __builtin_amdgcn_tensor_load_to_lds(g0, g1, z4, z4, 0);
#endif
}

// ---------------------------------------------------------------------------
// Kernel 1: h = silu([node_embed; ori] @ W1 + b1) * gate   -> bf16 [2E, H]
// Block: 256 threads (8 waves), tile 128(M) x 128(N), K-step 32.
// Wave (wid): waveM = wid&3 -> 32 rows (2 M-tiles), waveN = wid>>2 -> 64 cols
// ---------------------------------------------------------------------------
__global__ __launch_bounds__(256) void fc1_silu_gate_kernel(
    const float* __restrict__ node_embed,
    const float* __restrict__ ori,
    const float* __restrict__ x_edge_c,
    const float* __restrict__ W1,
    const float* __restrict__ b1,
    __bf16* __restrict__ h_out)
{
    __shared__ __bf16 As[128 * 32];
    __shared__ __bf16 Bs[128 * 32];
    unsigned int* As32 = (unsigned int*)As;
    unsigned int* Bs32 = (unsigned int*)Bs;

    const int tid   = threadIdx.x;
    const int lane  = tid & 31;
    const int wid   = tid >> 5;
    const int waveM = wid & 3;
    const int waveN = wid >> 2;
    const int rowBase = blockIdx.y * 128;   // over 2E
    const int colBase = blockIdx.x * 128;   // over H

    v8f acc[2][4];
    #pragma unroll
    for (int mi = 0; mi < 2; ++mi)
        #pragma unroll
        for (int ni = 0; ni < 4; ++ni)
            acc[mi][ni] = (v8f){};

    for (int k0 = 0; k0 < DD; k0 += 32) {
        // Stage A: 128 rows x 32 K (f32 -> bf16 packed pairs), K-contiguous
        #pragma unroll
        for (int i = 0; i < 8; ++i) {
            int idx = tid + i * 256;          // 2048 pair-slots
            int kp  = idx & 15;               // k-pair 0..15
            int m   = idx >> 4;               // row 0..127
            int r   = rowBase + m;
            int rc  = (r < TWO_E) ? r : (TWO_E - 1);
            const float* src = (rc < EN) ? (node_embed + (size_t)rc * DD)
                                         : (ori + (size_t)(rc - EN) * DD);
            float f0 = src[k0 + 2 * kp];
            float f1 = src[k0 + 2 * kp + 1];
            As32[m * 16 + kp] = pack2bf(f0, f1);
        }
        // Stage B: W1 tile [32 x 128] transposed to LDS [n][k]
        #pragma unroll
        for (int i = 0; i < 8; ++i) {
            int idx = tid + i * 256;
            int n   = idx & 127;
            int kp  = idx >> 7;
            size_t g = (size_t)(k0 + 2 * kp) * HH + colBase + n;
            Bs32[n * 16 + kp] = pack2bf(W1[g], W1[g + HH]);
        }
        __syncthreads();

        v16bf a[2], b[4];
        #pragma unroll
        for (int mi = 0; mi < 2; ++mi)
            a[mi] = frag_ld(As, waveM * 32 + mi * 16 + (lane & 15), lane);
        #pragma unroll
        for (int ni = 0; ni < 4; ++ni)
            b[ni] = frag_ld(Bs, waveN * 64 + ni * 16 + (lane & 15), lane);

        #pragma unroll
        for (int mi = 0; mi < 2; ++mi)
            #pragma unroll
            for (int ni = 0; ni < 4; ++ni)
                acc[mi][ni] = __builtin_amdgcn_wmma_f32_16x16x32_bf16(
                    false, a[mi], false, b[ni], (short)0, acc[mi][ni],
                    false, false);

        __syncthreads();
    }

    // Epilogue: bias + SiLU + edge gate, store bf16
    #pragma unroll
    for (int ni = 0; ni < 4; ++ni) {
        const int col = colBase + waveN * 64 + ni * 16 + (lane & 15);
        const float bias = b1[col];
        #pragma unroll
        for (int mi = 0; mi < 2; ++mi) {
            #pragma unroll
            for (int v = 0; v < 8; ++v) {
                int r = rowBase + waveM * 32 + mi * 16 + ((lane >> 4) << 3) + v;
                if (r < TWO_E) {
                    int gr = (r < EN) ? r : (r - EN);
                    float gate = x_edge_c[(size_t)gr * HH + col];
                    float val  = acc[mi][ni][v] + bias;
                    h_out[(size_t)r * HH + col] = (__bf16)(silu_f(val) * gate);
                }
            }
        }
    }
}

// ---------------------------------------------------------------------------
// Kernel 2: out[e] = silu(h[e] @ W2 + b2) + silu(h[e+E] @ W2 + b2)
// Block: 256 threads, tile 64 out-rows (=128 h-rows, both halves) x 128 cols.
// A tile (pure bf16 copy) is DMAed by the Tensor Data Mover: one
// TENSOR_LOAD_TO_LDS per concat half, issued by wave 0, TENSORcnt-waited.
// ---------------------------------------------------------------------------
__global__ __launch_bounds__(256) void fc2_silu_add_kernel(
    const __bf16* __restrict__ h,
    const float* __restrict__ W2,
    const float* __restrict__ b2,
    float* __restrict__ out)
{
    __shared__ __bf16 As[128 * 32];
    __shared__ __bf16 Bs[128 * 32];
    unsigned int* Bs32 = (unsigned int*)Bs;
    const char* hbytes = (const char*)h;     // row pitch = HH*2 = 512 bytes

    const int tid   = threadIdx.x;
    const int lane  = tid & 31;
    const int wid   = tid >> 5;
    const int waveM = wid & 3;
    const int waveN = wid >> 2;
    const int eBase   = blockIdx.y * 64;     // over E
    const int colBase = blockIdx.x * 128;    // over D

    const unsigned asbase = (unsigned)(uintptr_t)(&As[0]);
    const unsigned rows_valid =
        (unsigned)((EN - eBase) < 64 ? (EN - eBase) : 64);

    v8f c0[4], c1[4];
    #pragma unroll
    for (int ni = 0; ni < 4; ++ni) { c0[ni] = (v8f){}; c1[ni] = (v8f){}; }

    for (int k0 = 0; k0 < HH; k0 += 32) {
        // Stage A via TDM: rows 0..63 = h[eBase+m], 64..127 = h[E+eBase+m].
        // TDM issues once per wave (EXEC ignored) -> only wave 0 issues.
        if (wid == 0) {
            tdm_load_tile_64x32(
                asbase,
                hbytes + ((size_t)eBase * HH + k0) * 2, rows_valid);
            tdm_load_tile_64x32(
                asbase + 64 * 64,
                hbytes + (((size_t)EN + eBase) * HH + k0) * 2, rows_valid);
        }
        // Stage B: W2 tile [32 x 128] transposed to LDS [n][k] (f32->bf16)
        #pragma unroll
        for (int i = 0; i < 8; ++i) {
            int idx = tid + i * 256;
            int n   = idx & 127;
            int kp  = idx >> 7;
            size_t g = (size_t)(k0 + 2 * kp) * DD + colBase + n;
            Bs32[n * 16 + kp] = pack2bf(W2[g], W2[g + DD]);
        }
        // TENSORcnt is per-wave: wave 0 waits for its DMAs, then the
        // workgroup barrier publishes the LDS tile to everyone.
        if (wid == 0)
            __builtin_amdgcn_s_wait_tensorcnt(0);
        __syncthreads();

        v16bf a0 = frag_ld(As, waveM * 16 + (lane & 15), lane);
        v16bf a1 = frag_ld(As, 64 + waveM * 16 + (lane & 15), lane);
        #pragma unroll
        for (int ni = 0; ni < 4; ++ni) {
            v16bf b = frag_ld(Bs, waveN * 64 + ni * 16 + (lane & 15), lane);
            c0[ni] = __builtin_amdgcn_wmma_f32_16x16x32_bf16(
                false, a0, false, b, (short)0, c0[ni], false, false);
            c1[ni] = __builtin_amdgcn_wmma_f32_16x16x32_bf16(
                false, a1, false, b, (short)0, c1[ni], false, false);
        }
        __syncthreads();
    }

    // Epilogue: silu(c0 + b2) + silu(c1 + b2), single fp32 store
    #pragma unroll
    for (int ni = 0; ni < 4; ++ni) {
        const int col = colBase + waveN * 64 + ni * 16 + (lane & 15);
        const float bias = b2[col];
        #pragma unroll
        for (int v = 0; v < 8; ++v) {
            int r = eBase + waveM * 16 + ((lane >> 4) << 3) + v;
            if (r < EN) {
                float y0 = silu_f(c0[ni][v] + bias);
                float y1 = silu_f(c1[ni][v] + bias);
                out[(size_t)r * DD + col] = y0 + y1;
            }
        }
    }
}

extern "C" void kernel_launch(void* const* d_in, const int* in_sizes, int n_in,
                              void* d_out, int out_size, void* d_ws, size_t ws_size,
                              hipStream_t stream) {
    (void)in_sizes; (void)n_in; (void)out_size; (void)ws_size;
    const float* node_embed = (const float*)d_in[0];
    const float* x_edge_c   = (const float*)d_in[1];
    const float* ori        = (const float*)d_in[2];
    const float* W1         = (const float*)d_in[3];
    const float* b1         = (const float*)d_in[4];
    const float* W2         = (const float*)d_in[5];
    const float* b2         = (const float*)d_in[6];
    // d_in[7] = edge_num (compile-time constant EN here)

    __bf16* h_ws = (__bf16*)d_ws;           // 2E * H bf16 = ~12.3 MB
    float*  out  = (float*)d_out;           // E * D fp32

    dim3 g1(HH / 128, (TWO_E + 127) / 128);             // (2, 188)
    fc1_silu_gate_kernel<<<g1, 256, 0, stream>>>(node_embed, ori, x_edge_c,
                                                 W1, b1, h_ws);

    dim3 g2(DD / 128, (EN + 63) / 64);                  // (49, 188)
    fc2_silu_add_kernel<<<g2, 256, 0, stream>>>(h_ws, W2, b2, out);
}
